// Transformer_84482006713113
// MI455X (gfx1250) — matrix operand order repaired and checked by reading, not compiled, not run on hardware
//
#include <hip/hip_runtime.h>
#include <math.h>

// ---------------- CDNA5 WMMA types ----------------
typedef __attribute__((ext_vector_type(16))) __bf16 v16bf;
typedef __attribute__((ext_vector_type(8)))  float  v8f;
typedef __attribute__((ext_vector_type(4)))  int    v4i;
typedef __attribute__((address_space(1))) v4i gv4i;   // global int4
typedef __attribute__((address_space(3))) v4i lv4i;   // LDS int4

#define HAS_ASYNC_LDS __has_builtin(__builtin_amdgcn_global_load_async_to_lds_b128)

static __device__ __forceinline__ __bf16 f2bf(float f) {
  unsigned u = __builtin_bit_cast(unsigned, f);
  unsigned r = u + 0x7FFFu + ((u >> 16) & 1u);   // round-to-nearest-even
  unsigned short h = (unsigned short)(r >> 16);
  return __builtin_bit_cast(__bf16, h);
}

static __device__ __forceinline__ float gelu_exact(float x) {
  return 0.5f * x * (1.0f + erff(x * 0.70710678118654752f));
}

// ---------------- problem constants ----------------
#define BB     8
#define CC     256
#define HH     64
#define WW     64
#define HWSZ   4096
#define GG     4
#define CGC    64
#define HEADS  8
#define DHH    32
#define D2C    1024
#define NPTS   16
#define BGC    32
#define SCALEA 0.17677669529663687f   // 32^-0.5

// GEMM block tile
#define BM 64
#define BN 128
#define BK 32
#define LDP (BK + 8)   // padded LDS row (40 bf16 = 80B; 16B-aligned chunks)

// =====================================================================
// LDS-tiled GEMM on pre-converted bf16 operands:
//   Y[b,m,n] = sum_k W[m,k] * X[b,k,n] + bias[m]
// X: [Bn,K,N] bf16, W: [M,K] bf16, Y: [Bn,M,N] f32.
// 256 threads / 8 waves; block tile 64M x 128N; K step 32.
// A panel staged via async global->LDS b128 copies (ASYNCcnt) when the
// builtin exists; B panel transpose-staged with packed b64 LDS stores.
// 4 WMMAs per wave per K step share one A fragment.
// =====================================================================
__global__ void __launch_bounds__(256)
gemm_bf16_wmma_lds(const __bf16* __restrict__ X,
                   const __bf16* __restrict__ Wt,
                   const float* __restrict__ bias,
                   float* __restrict__ Y,
                   int M, int N, int K, int mblocks, int nblocks)
{
  int blk = blockIdx.x;
  int nb = blk % nblocks;
  int mb = (blk / nblocks) % mblocks;
  int b  = blk / (nblocks * mblocks);
  int tid = threadIdx.x;
  int wave = tid >> 5, lane = tid & 31;
  int row = lane & 15, hi = lane >> 4;
  int mtl = wave & 3;          // local M tile (4 x 16 rows)
  int ng  = wave >> 2;         // N group (2 x 64 cols)

  __shared__ __bf16 As[BM][LDP];   // [m][k]
  __shared__ __bf16 Bs[BN][LDP];   // [n][k]  (transposed for contiguous frag reads)

  const __bf16* Wp = Wt + (size_t)(mb * BM) * K;
  const __bf16* Xp = X + (size_t)b * K * N + (size_t)(nb * BN);

  v8f acc[4] = {};

  for (int k0 = 0; k0 < K; k0 += BK) {
    if (k0 + BK < K)
      __builtin_prefetch(Xp + (size_t)(k0 + BK + (tid >> 5)) * N, 0, 1);

    // ---- stage A panel: 64 rows x 32 k bf16 = 256 x 16B, one chunk/thread
    {
      int r  = tid >> 2;          // 0..63
      int c8 = (tid & 3) * 8;     // 0,8,16,24
#if HAS_ASYNC_LDS
      __builtin_amdgcn_global_load_async_to_lds_b128(
          (gv4i*)(Wp + (size_t)r * K + k0 + c8),
          (lv4i*)(&As[r][c8]),
          0, 0);
#else
      *(uint4*)&As[r][c8] = *(const uint4*)(Wp + (size_t)r * K + k0 + c8);
#endif
    }
    // ---- stage B panel (transposed): 1024 slots of 4 K-elems, packed b64
#pragma unroll
    for (int i = 0; i < 4; ++i) {
      int q  = tid + i * 256;     // 0..1023
      int n  = q & 127;
      int kq = (q >> 7) * 4;      // 0..28
      const __bf16* src = Xp + (size_t)(k0 + kq) * N + n;
      unsigned e0 = *(const unsigned short*)(src);
      unsigned e1 = *(const unsigned short*)(src + N);
      unsigned e2 = *(const unsigned short*)(src + 2 * (size_t)N);
      unsigned e3 = *(const unsigned short*)(src + 3 * (size_t)N);
      uint2 pk;
      pk.x = e0 | (e1 << 16);
      pk.y = e2 | (e3 << 16);
      *(uint2*)&Bs[n][kq] = pk;
    }
#if HAS_ASYNC_LDS
#if __has_builtin(__builtin_amdgcn_s_wait_asynccnt)
    __builtin_amdgcn_s_wait_asynccnt(0);
#else
    asm volatile("s_wait_asynccnt 0" ::: "memory");
#endif
#endif
    __syncthreads();

    // ---- A fragment (contiguous K order, matching B): lanes 0-15 hold K 0-15,
    //      lanes 16-31 hold K 16-31; contiguous in LDS.
    v16bf a;
    {
      const __bf16* ap = &As[mtl * 16 + row][0];
#pragma unroll
      for (int h = 0; h < 16; ++h) a[h] = ap[16 * hi + h];
    }
    // ---- 4 B fragments + 4 WMMAs sharing the A fragment
#pragma unroll
    for (int t = 0; t < 4; ++t) {
      const __bf16* bp = &Bs[ng * 64 + t * 16 + row][0];
      v16bf bm;
#pragma unroll
      for (int h = 0; h < 16; ++h) bm[h] = bp[16 * hi + h];
      acc[t] = __builtin_amdgcn_wmma_f32_16x16x32_bf16(
          false, a, false, bm, (short)0, acc[t], false, false);
    }
    __syncthreads();
  }

  // ---- store: VGPR v, lane l -> M = v + 8*(l>=16), N = l&15
  int m0 = mb * BM + mtl * 16;
  int n0 = nb * BN + ng * 64;
#pragma unroll
  for (int t = 0; t < 4; ++t) {
    int n = n0 + t * 16 + row;
#pragma unroll
    for (int v = 0; v < 8; ++v) {
      int m = m0 + v + 8 * hi;
      Y[((size_t)b * M + m) * N + n] = acc[t][v] + bias[m];
    }
  }
}

// =====================================================================
// Direct-load WMMA GEMM for small N (K/V over the 16 sampled points).
// f32 inputs, converts in registers. One wave per 16x16 tile.
// =====================================================================
__global__ void gemm_bf16_wmma_small(const float* __restrict__ X,
                                     const float* __restrict__ Wt,
                                     const float* __restrict__ bias,
                                     float* __restrict__ Y,
                                     int Bn, int M, int N, int K)
{
  int wave = blockIdx.x * 8 + (threadIdx.x >> 5);
  int lane = threadIdx.x & 31;
  int ntiles = N >> 4, mtiles = M >> 4;
  int total = Bn * mtiles * ntiles;
  if (wave >= total) return;                 // wave-uniform guard
  int nt = wave % ntiles;
  int mt = (wave / ntiles) % mtiles;
  int b  = wave / (ntiles * mtiles);
  int m0 = mt << 4, n0 = nt << 4;
  int row = lane & 15;
  int hi  = lane >> 4;

  const float* Xb = X + (size_t)b * K * N;
  v8f acc = {};
  for (int k0 = 0; k0 < K; k0 += 32) {
    v16bf a, bm;
#pragma unroll
    for (int h = 0; h < 16; ++h) {
      int ka = k0 + h + 16 * hi;
      a[h] = f2bf(Wt[(size_t)(m0 + row) * K + ka]);
    }
#pragma unroll
    for (int h = 0; h < 16; ++h) {
      int kb = k0 + h + 16 * hi;
      bm[h] = f2bf(Xb[(size_t)kb * N + (n0 + row)]);
    }
    acc = __builtin_amdgcn_wmma_f32_16x16x32_bf16(
        false, a, false, bm, (short)0, acc, false, false);
  }
  int n = n0 + row;
#pragma unroll
  for (int v = 0; v < 8; ++v) {
    int m = m0 + v + 8 * hi;
    Y[((size_t)b * M + m) * N + n] = acc[v] + bias[m];
  }
}

// =====================================================================
// Channel LayerNorm over NCHW; emits f32 (for sampling path) and bf16
// (GEMM operand) in one pass.
// =====================================================================
__global__ void ln_channel(const float* __restrict__ x, const float* __restrict__ g,
                           const float* __restrict__ bta,
                           float* __restrict__ y32, __bf16* __restrict__ y16,
                           int C, int total_pix)
{
  int idx = blockIdx.x * blockDim.x + threadIdx.x;
  if (idx >= total_pix) return;
  int b = idx / HWSZ, hw = idx - b * HWSZ;
  const float* xp = x + (size_t)b * C * HWSZ + hw;
  float s = 0.f, s2 = 0.f;
  for (int c = 0; c < C; ++c) { float v = xp[(size_t)c * HWSZ]; s += v; s2 += v * v; }
  float m = s / (float)C;
  float var = s2 / (float)C - m * m;
  float r = rsqrtf(var + 1e-5f);
  float* yp = y32 + (size_t)b * C * HWSZ + hw;
  __bf16* hp = y16 + (size_t)b * C * HWSZ + hw;
  for (int c = 0; c < C; ++c) {
    float v = (xp[(size_t)c * HWSZ] - m) * r * g[c] + bta[c];
    yp[(size_t)c * HWSZ] = v;
    hp[(size_t)c * HWSZ] = f2bf(v);
  }
}

// =====================================================================
// Offset net stage 1: depthwise 16x16 conv, stride 16, pad 0
// =====================================================================
__global__ void offset_dwconv(const float* __restrict__ q,
                              const float* __restrict__ w,   // [64,256]
                              const float* __restrict__ bias,
                              float* __restrict__ o)         // [32,64,16]
{
  int idx = blockIdx.x * blockDim.x + threadIdx.x;
  if (idx >= BGC * CGC * NPTS) return;
  int p  = idx & 15;
  int c  = (idx >> 4) & 63;
  int bg = idx >> 10;
  int b = bg >> 2, g = bg & 3;
  int hy = p >> 2, hx = p & 3;
  const float* base = q + ((size_t)b * CC + g * CGC + c) * HWSZ + (hy * 16) * WW + hx * 16;
  const float* wp = w + c * 256;
  float acc = bias[c];
  for (int ky = 0; ky < 16; ++ky)
#pragma unroll
    for (int kx = 0; kx < 16; ++kx)
      acc += base[ky * WW + kx] * wp[ky * 16 + kx];
  o[idx] = acc;
}

// =====================================================================
// Offset net stage 2: LN(64ch) -> GELU -> 1x1 to 2 -> tanh*range + ref
// =====================================================================
__global__ void offset_finalize(const float* __restrict__ o,
                                const float* __restrict__ lng,
                                const float* __restrict__ lnb,
                                const float* __restrict__ ow,  // [2,64]
                                float* __restrict__ pos)       // [32,16,2] (y,x)
{
  int idx = blockIdx.x * blockDim.x + threadIdx.x;
  if (idx >= BGC * NPTS) return;
  int p  = idx & 15;
  int bg = idx >> 4;
  const float* op = o + (size_t)bg * (CGC * NPTS) + p;
  float s = 0.f, s2 = 0.f;
  for (int c = 0; c < CGC; ++c) { float v = op[c * NPTS]; s += v; s2 += v * v; }
  float m = s / (float)CGC;
  float var = s2 / (float)CGC - m * m;
  float r = rsqrtf(var + 1e-5f);
  float a0 = 0.f, a1 = 0.f;
  for (int c = 0; c < CGC; ++c) {
    float v = (op[c * NPTS] - m) * r * lng[c] + lnb[c];
    float gl = gelu_exact(v);
    a0 += ow[c] * gl;        // dy channel
    a1 += ow[CGC + c] * gl;  // dx channel
  }
  int hy = p >> 2, hx = p & 3;
  float ry = ((hy + 0.5f) / 4.f) * 2.f - 1.f;
  float rx = ((hx + 0.5f) / 4.f) * 2.f - 1.f;
  float dy = tanhf(a0) * (16.0f / 3.0f);   // ORF / (Hk-1)
  float dx = tanhf(a1) * (16.0f / 3.0f);
  pos[idx * 2 + 0] = dy + ry;
  pos[idx * 2 + 1] = dx + rx;
}

// =====================================================================
// Bilinear grid sample (align_corners=True, zero pad)
// =====================================================================
__global__ void grid_sample_k(const float* __restrict__ img,  // [B,256,4096]
                              const float* __restrict__ pos,  // [32,16,2]
                              float* __restrict__ xs)         // [B,256,16]
{
  int idx = blockIdx.x * blockDim.x + threadIdx.x;
  if (idx >= BB * CC * NPTS) return;
  int n  = idx & 15;
  int ch = (idx >> 4) & 255;
  int b  = idx >> 12;
  int g  = ch >> 6;
  int bg = b * GG + g;
  float gy = pos[(bg * NPTS + n) * 2 + 0];
  float gx = pos[(bg * NPTS + n) * 2 + 1];
  float fx = (gx + 1.f) * 31.5f;
  float fy = (gy + 1.f) * 31.5f;
  float x0 = floorf(fx), y0 = floorf(fy);
  float wx = fx - x0, wy = fy - y0;
  const float* ip = img + ((size_t)b * CC + ch) * HWSZ;
  float acc = 0.f;
#pragma unroll
  for (int t = 0; t < 4; ++t) {
    float xi = x0 + (float)(t & 1);
    float yi = y0 + (float)(t >> 1);
    float wgt = ((t & 1) ? wx : 1.f - wx) * ((t >> 1) ? wy : 1.f - wy);
    if (xi >= 0.f && xi <= 63.f && yi >= 0.f && yi <= 63.f) {
      int xc = (int)xi, yc = (int)yi;
      acc += ip[yc * WW + xc] * wgt;
    }
  }
  xs[idx] = acc;
}

// =====================================================================
// Attention: per (b,head,query) softmax over 16 sampled keys.
// Output bf16 (feeds the projection GEMM).
// =====================================================================
__global__ void attn_k(const float* __restrict__ q,  // [B,256,4096]
                       const float* __restrict__ k,  // [B,256,16]
                       const float* __restrict__ v,  // [B,256,16]
                       __bf16* __restrict__ out)     // [B,256,4096] bf16
{
  int gid = blockIdx.x * blockDim.x + threadIdx.x;
  if (gid >= BB * HEADS * HWSZ) return;
  int m  = gid & (HWSZ - 1);
  int bh = gid >> 12;
  int b = bh >> 3, hh = bh & 7;
  const float* qp = q + ((size_t)b * CC + hh * DHH) * HWSZ + m;
  const float* kp = k + ((size_t)b * CC + hh * DHH) * NPTS;
  const float* vp = v + ((size_t)b * CC + hh * DHH) * NPTS;
  float qr[DHH];
#pragma unroll
  for (int c = 0; c < DHH; ++c) qr[c] = qp[(size_t)c * HWSZ];
  float lg[NPTS];
  float mx = -3.4e38f;
#pragma unroll
  for (int n = 0; n < NPTS; ++n) {
    float s = 0.f;
#pragma unroll
    for (int c = 0; c < DHH; ++c) s += qr[c] * kp[c * NPTS + n];
    s *= SCALEA;
    lg[n] = s;
    mx = fmaxf(mx, s);
  }
  float den = 0.f;
#pragma unroll
  for (int n = 0; n < NPTS; ++n) { lg[n] = expf(lg[n] - mx); den += lg[n]; }
  float rd = 1.f / den;
  __bf16* op = out + ((size_t)b * CC + hh * DHH) * HWSZ + m;
#pragma unroll
  for (int c = 0; c < DHH; ++c) {
    float s = 0.f;
#pragma unroll
    for (int n = 0; n < NPTS; ++n) s += lg[n] * vp[c * NPTS + n];
    op[(size_t)c * HWSZ] = f2bf(s * rd);
  }
}

// =====================================================================
// x += src * layer_scale[c]
// =====================================================================
__global__ void resid_scale(float* __restrict__ x, const float* __restrict__ src,
                            const float* __restrict__ ls, int C, int total)
{
  int idx = blockIdx.x * blockDim.x + threadIdx.x;
  if (idx >= total) return;
  int c = (idx >> 12) % C;
  x[idx] = src[idx] * ls[c] + x[idx];
}

// =====================================================================
// depthwise 3x3 (pad 1) + residual + exact GELU; bf16 out for GEMM2
// =====================================================================
__global__ void dwconv3_gelu(const float* __restrict__ h,  // [B,1024,4096]
                             const float* __restrict__ w,  // [1024,9]
                             const float* __restrict__ bias,
                             __bf16* __restrict__ o, int total)
{
  int idx = blockIdx.x * blockDim.x + threadIdx.x;
  if (idx >= total) return;
  int hw = idx & (HWSZ - 1);
  int c  = (idx >> 12) & (D2C - 1);
  int yy = hw >> 6, xx = hw & 63;
  const float* hp = h + (size_t)(idx - hw);
  const float* wp = w + c * 9;
  float acc = bias[c];
#pragma unroll
  for (int dy = -1; dy <= 1; ++dy)
#pragma unroll
    for (int dx = -1; dx <= 1; ++dx) {
      int y2 = yy + dy, x2 = xx + dx;
      if (y2 >= 0 && y2 < HH && x2 >= 0 && x2 < WW)
        acc += hp[y2 * WW + x2] * wp[(dy + 1) * 3 + (dx + 1)];
    }
  float val = h[idx] + acc;
  o[idx] = f2bf(gelu_exact(val));
}

__global__ void cvt_bf16(const float* __restrict__ s, __bf16* __restrict__ d, int n)
{
  int i = blockIdx.x * blockDim.x + threadIdx.x;
  if (i < n) d[i] = f2bf(s[i]);
}

__global__ void copy_k(const float* __restrict__ s, float* __restrict__ d, int n)
{
  int i = blockIdx.x * blockDim.x + threadIdx.x;
  if (i < n) d[i] = s[i];
}

// =====================================================================
// host-side orchestration
// =====================================================================
extern "C" void kernel_launch(void* const* d_in, const int* in_sizes, int n_in,
                              void* d_out, int out_size, void* d_ws, size_t ws_size,
                              hipStream_t stream) {
  (void)in_sizes; (void)n_in; (void)out_size; (void)ws_size;

  const float* x         = (const float*)d_in[0];
  const float* ln_attn_g = (const float*)d_in[1];
  const float* ln_attn_b = (const float*)d_in[2];
  const float* wq        = (const float*)d_in[3];
  const float* bq        = (const float*)d_in[4];
  const float* wk        = (const float*)d_in[5];
  const float* bk        = (const float*)d_in[6];
  const float* wv        = (const float*)d_in[7];
  const float* bv        = (const float*)d_in[8];
  const float* wo        = (const float*)d_in[9];
  const float* bo        = (const float*)d_in[10];
  const float* odw_w     = (const float*)d_in[11];
  const float* odw_b     = (const float*)d_in[12];
  const float* oln_g     = (const float*)d_in[13];
  const float* oln_b     = (const float*)d_in[14];
  const float* oww       = (const float*)d_in[15];
  const float* ls_attn   = (const float*)d_in[16];
  const float* ln_mlp_g  = (const float*)d_in[17];
  const float* ln_mlp_b  = (const float*)d_in[18];
  const float* w1        = (const float*)d_in[19];
  const float* b1        = (const float*)d_in[20];
  const float* dwc_w     = (const float*)d_in[21];
  const float* dwc_b     = (const float*)d_in[22];
  const float* w2        = (const float*)d_in[23];
  const float* b2        = (const float*)d_in[24];
  const float* ls_mlp    = (const float*)d_in[25];

  const size_t NACT = (size_t)BB * CC * HWSZ;        // 8.39M
  const size_t NHID = (size_t)BB * D2C * HWSZ;       // 33.5M
  const int NWQ = 2 * CC * CC;                       // 131072
  const int NW1 = 2 * D2C * CC;                      // 524288

  // f32 scratch region
  float* wsf = (float*)d_ws;
  size_t off = 0;
  float* XCUR = wsf + off; off += NACT;
  float* YBUF = wsf + off; off += NACT;
  float* QBUF = wsf + off; off += NACT;
  float* H1   = wsf + off; off += NHID;
  float* OOFF = wsf + off; off += (size_t)BGC * CGC * NPTS;
  float* POS  = wsf + off; off += (size_t)BGC * NPTS * 2;
  float* XS   = wsf + off; off += (size_t)BB * CC * NPTS;
  float* KB   = wsf + off; off += (size_t)BB * CC * NPTS;
  float* VB   = wsf + off; off += (size_t)BB * CC * NPTS;
  // bf16 scratch region
  __bf16* wsb = (__bf16*)(wsf + off);
  size_t ob = 0;
  __bf16* Y16  = wsb + ob; ob += NACT;
  __bf16* A16  = wsb + ob; ob += NACT;
  __bf16* H216 = wsb + ob; ob += NHID;
  __bf16* WQ16 = wsb + ob; ob += NWQ;
  __bf16* WO16 = wsb + ob; ob += NWQ;
  __bf16* W116 = wsb + ob; ob += NW1;
  __bf16* W216 = wsb + ob; ob += NW1;

  const int nAct = (int)NACT, nHid = (int)NHID, nPix = BB * HWSZ;

  auto gemm_big = [&](const __bf16* X, const __bf16* Wt, const float* bias, float* Y,
                      int Bn, int M, int N, int K) {
    int mbl = M / BM, nbl = N / BN;
    gemm_bf16_wmma_lds<<<Bn * mbl * nbl, 256, 0, stream>>>(X, Wt, bias, Y, M, N, K, mbl, nbl);
  };
  auto gemm_small = [&](const float* X, const float* Wt, const float* bias, float* Y,
                        int Bn, int M, int N, int K) {
    int tiles = Bn * (M >> 4) * (N >> 4);
    gemm_bf16_wmma_small<<<(tiles + 7) / 8, 256, 0, stream>>>(X, Wt, bias, Y, Bn, M, N, K);
  };

  // one-time weight conversion (both layers at once)
  cvt_bf16<<<(NWQ + 255) / 256, 256, 0, stream>>>(wq, WQ16, NWQ);
  cvt_bf16<<<(NWQ + 255) / 256, 256, 0, stream>>>(wo, WO16, NWQ);
  cvt_bf16<<<(NW1 + 255) / 256, 256, 0, stream>>>(w1, W116, NW1);
  cvt_bf16<<<(NW1 + 255) / 256, 256, 0, stream>>>(w2, W216, NW1);

  copy_k<<<(nAct + 255) / 256, 256, 0, stream>>>(x, XCUR, nAct);

  for (int l = 0; l < 2; ++l) {
    const float* lag = ln_attn_g + l * CC;  const float* lab = ln_attn_b + l * CC;
    const __bf16* wql = WQ16 + (size_t)l * CC * CC;  const float* bql = bq + l * CC;
    const float* wkl = wk + (size_t)l * CC * CC;  const float* bkl = bk + l * CC;
    const float* wvl = wv + (size_t)l * CC * CC;  const float* bvl = bv + l * CC;
    const __bf16* wol = WO16 + (size_t)l * CC * CC;  const float* bol = bo + l * CC;
    const float* odwl = odw_w + (size_t)l * CGC * 256;
    const float* odbl = odw_b + l * CGC;
    const float* olgl = oln_g + l * CGC;  const float* olbl = oln_b + l * CGC;
    const float* owl  = oww + l * 2 * CGC;
    const float* lsa  = ls_attn + l * CC;
    const float* lmg = ln_mlp_g + l * CC; const float* lmb = ln_mlp_b + l * CC;
    const __bf16* w1l = W116 + (size_t)l * D2C * CC;  const float* b1l = b1 + l * D2C;
    const float* dwl = dwc_w + (size_t)l * D2C * 9; const float* dbl = dwc_b + l * D2C;
    const __bf16* w2l = W216 + (size_t)l * CC * D2C;  const float* b2l = b2 + l * CC;
    const float* lsm = ls_mlp + l * CC;

    // ---- attention branch ----
    ln_channel<<<(nPix + 255) / 256, 256, 0, stream>>>(XCUR, lag, lab, YBUF, Y16, CC, nPix);
    gemm_big(Y16, wql, bql, QBUF, BB, CC, HWSZ, CC);                   // Q
    offset_dwconv<<<(BGC * CGC * NPTS + 255) / 256, 256, 0, stream>>>(QBUF, odwl, odbl, OOFF);
    offset_finalize<<<(BGC * NPTS + 255) / 256, 256, 0, stream>>>(OOFF, olgl, olbl, owl, POS);
    grid_sample_k<<<(BB * CC * NPTS + 255) / 256, 256, 0, stream>>>(YBUF, POS, XS);
    gemm_small(XS, wkl, bkl, KB, BB, CC, NPTS, CC);                    // K
    gemm_small(XS, wvl, bvl, VB, BB, CC, NPTS, CC);                    // V
    attn_k<<<(BB * HEADS * HWSZ + 255) / 256, 256, 0, stream>>>(QBUF, KB, VB, A16);
    gemm_big(A16, wol, bol, YBUF, BB, CC, HWSZ, CC);                   // out proj
    resid_scale<<<(nAct + 255) / 256, 256, 0, stream>>>(XCUR, YBUF, lsa, CC, nAct);

    // ---- MLP branch ----
    ln_channel<<<(nPix + 255) / 256, 256, 0, stream>>>(XCUR, lmg, lmb, YBUF, Y16, CC, nPix);
    gemm_big(Y16, w1l, b1l, H1, BB, D2C, HWSZ, CC);                    // 256 -> 1024
    dwconv3_gelu<<<(nHid + 255) / 256, 256, 0, stream>>>(H1, dwl, dbl, H216, nHid);
    gemm_big(H216, w2l, b2l, YBUF, BB, CC, HWSZ, D2C);                 // 1024 -> 256
    resid_scale<<<(nAct + 255) / 256, 256, 0, stream>>>(XCUR, YBUF, lsm, CC, nAct);
  }

  copy_k<<<(nAct + 255) / 256, 256, 0, stream>>>(XCUR, (float*)d_out, nAct);
}
